// MultiHeadAttention_7284264534284
// MI455X (gfx1250) — matrix operand order repaired and checked by reading, not compiled, not run on hardware
//
#include <hip/hip_runtime.h>
#include <hip/hip_bf16.h>
#include <stdint.h>

// ---------------------------------------------------------------------------
// MI455X (gfx1250) multi-head attention, flash-style, bf16 WMMA / f32 accum.
//   N=4096 tokens, D=2048 model dim, H=16 heads, DK=128 head dim.
// ~2.4e11 FLOP vs ~120 MB HBM traffic -> matrix-compute bound; bf16 working
// set (~92 MB) is L2-resident (192 MB), so both GEMM phases stream from L2.
// Round-4: double buffer rewritten with compile-time stage variables (no
// dynamic array indexing) so SROA keeps all fragments in VGPRs -- the dynamic
// ping-pong index was forcing the whole buffer into scratch/LDS spill.
// ---------------------------------------------------------------------------

typedef __attribute__((ext_vector_type(16))) __bf16        v16bf;
typedef __attribute__((ext_vector_type(8)))  float         v8f;
typedef __attribute__((ext_vector_type(8)))  unsigned short us8;

union Frag { v16bf bf; us8 u[2]; };

constexpr int N_ = 4096, D_ = 2048, H_ = 16, DK_ = 128;

__device__ __forceinline__ unsigned short f2bf(float f) {
  unsigned int u = __float_as_uint(f);
  u += 0x7FFFu + ((u >> 16) & 1u);        // round-to-nearest-even
  return (unsigned short)(u >> 16);
}

__device__ __forceinline__ v8f wmma_bf16(v16bf a, v16bf b, v8f c) {
  // (neg_a, A, neg_b, B, c_mod, C, reuse_a, reuse_b)
  return __builtin_amdgcn_wmma_f32_16x16x32_bf16(false, a, false, b,
                                                 (short)0, c, false, false);
}

// ------------------------- conversion kernels ------------------------------

// 8 elements / thread: two b128 loads, one b128 store.
__global__ void cvt_bf16_kernel(const float* __restrict__ src,
                                unsigned short* __restrict__ dst, int n8) {
  int i = blockIdx.x * blockDim.x + threadIdx.x;
  if (i >= n8) return;
  size_t base = (size_t)i * 8;
  us8 p;
#pragma unroll
  for (int j = 0; j < 8; ++j) p[j] = f2bf(src[base + j]);
  *(us8*)(dst + base) = p;
}

// w: [H][D][DK] f32 -> wt: [H][DK][D] bf16, coalesced both ways via LDS tile.
// grid (D/32, DK/32, H), block (32, 8).
__global__ void wt_transpose_kernel(const float* __restrict__ w,
                                    unsigned short* __restrict__ wt) {
  __shared__ unsigned short tile[32][34];   // pad 34: distinct dword banks
  const int d0 = blockIdx.x * 32, k0 = blockIdx.y * 32, h = blockIdx.z;
#pragma unroll
  for (int rr = 0; rr < 32; rr += 8) {
    int d = d0 + threadIdx.y + rr;
    int k = k0 + threadIdx.x;
    tile[threadIdx.y + rr][threadIdx.x] = f2bf(w[((size_t)h * D_ + d) * DK_ + k]);
  }
  __syncthreads();
#pragma unroll
  for (int rr = 0; rr < 32; rr += 8) {
    int k = k0 + threadIdx.y + rr;
    int d = d0 + threadIdx.x;
    wt[((size_t)h * DK_ + k) * D_ + d] = tile[threadIdx.x][threadIdx.y + rr];
  }
}

// ------------------------- QKV projection GEMM -----------------------------

__device__ __forceinline__ void load_stage(Frag& a, Frag (&b)[4],
                                           const unsigned short* aBase,
                                           const unsigned short* const (&bp)[4],
                                           int d0) {
  a.u[0] = *(const us8*)(aBase + d0);
  a.u[1] = *(const us8*)(aBase + d0 + 8);
#pragma unroll
  for (int t = 0; t < 4; ++t) {
    b[t].u[0] = *(const us8*)(bp[t] + d0);
    b[t].u[1] = *(const us8*)(bp[t] + d0 + 8);
  }
}

__device__ __forceinline__ void wmma4(v8f (&acc)[4], const Frag& a,
                                      const Frag (&b)[4]) {
#pragma unroll
  for (int t = 0; t < 4; ++t) acc[t] = wmma_bf16(a.bf, b[t].bf, acc[t]);
}

// grid (N/32, H, 3), block 128 (4 waves -> 1 wave/SIMD, full VGPR budget).
// Wave = 16 rows x 64 cols:
//   rows = blockIdx.x*32 + (wid&1)*16
//   cols = (wid>>1)*64 + t*16, t in 0..3
// Double-buffered with compile-time stage variables (no dynamic indexing).
// z=0->Q, z=1->K ([H][N][DK]), z=2->V (transposed [H][DK][N], packed stores).

__global__ void __launch_bounds__(128)
qkv_proj_kernel(const unsigned short* __restrict__ tok,
                const unsigned short* __restrict__ wqT,
                const unsigned short* __restrict__ wkT,
                const unsigned short* __restrict__ wvT,
                unsigned short* __restrict__ Qb,
                unsigned short* __restrict__ Kb,
                unsigned short* __restrict__ VtB) {
  const int z    = blockIdx.z;
  const int h    = blockIdx.y;
  const int lane = threadIdx.x & 31;
  const int wid  = threadIdx.x >> 5;
  const int m0   = blockIdx.x * 32 + (wid & 1) * 16;
  const int c0   = (wid >> 1) * 64;
  const int ln16 = lane & 15;
  const int half8 = (lane & 16) ? 8 : 0;
  const int kbA   = (lane & 16) ? 16 : 0;   // A-frag K base
  const int kbB   = (lane & 16) ? 16 : 0;   // B-frag K base

  const unsigned short* wt = (z == 0) ? wqT : (z == 1) ? wkT : wvT;
  unsigned short* out      = (z == 0) ? Qb  : (z == 1) ? Kb  : VtB;

  const unsigned short* aBase = tok + (size_t)(m0 + ln16) * D_ + kbA;
  const unsigned short* bp[4];
#pragma unroll
  for (int t = 0; t < 4; ++t)
    bp[t] = wt + ((size_t)h * DK_ + c0 + t * 16 + ln16) * D_ + kbB;

  v8f acc[4];
#pragma unroll
  for (int t = 0; t < 4; ++t) acc[t] = (v8f){};

  Frag a0, a1;
  Frag b0[4], b1[4];

  load_stage(a0, b0, aBase, bp, 0);

  // D_ % 64 == 0: two pipeline stages per iteration, constant stage names.
  for (int d0 = 0; d0 < D_; d0 += 64) {
    load_stage(a1, b1, aBase, bp, d0 + 32);   // always in range
    wmma4(acc, a0, b0);
    if (d0 + 64 < D_) load_stage(a0, b0, aBase, bp, d0 + 64);
    wmma4(acc, a1, b1);
  }

  if (z < 2) {
    // [H][N][DK]: lanes 0-15 write 32 contiguous bytes per (t,r)
#pragma unroll
    for (int t = 0; t < 4; ++t) {
#pragma unroll
      for (int r = 0; r < 8; ++r) {
        int orow = m0 + r + half8;
        int ocol = c0 + t * 16 + ln16;
        out[((size_t)h * N_ + orow) * DK_ + ocol] = f2bf(acc[t][r]);
      }
    }
  } else {
    // [H][DK][N]: per lane, r=0..7 are 8 consecutive tokens -> one b128 store
#pragma unroll
    for (int t = 0; t < 4; ++t) {
      us8 pk;
#pragma unroll
      for (int r = 0; r < 8; ++r) pk[r] = f2bf(acc[t][r]);
      int ocol = c0 + t * 16 + ln16;
      *(us8*)(out + ((size_t)h * DK_ + ocol) * N_ + m0 + half8) = pk;
    }
  }
}

// ------------------------- fused flash attention ---------------------------
// grid (N/64, H), block 128 (4 waves). Each wave: 16 query rows, online
// softmax over all 4096 keys in tiles of 32. V-fragment loads are split 4+4:
// the first four overlap the softmax VALU, the last four overlap the LDS
// P-bounce and the first P.V WMMAs.

__global__ void __launch_bounds__(128)
attention_kernel(const unsigned short* __restrict__ Qb,
                 const unsigned short* __restrict__ Kb,
                 const unsigned short* __restrict__ VtB,
                 float* __restrict__ out) {
  __shared__ __align__(16) unsigned short pshm[4][16 * 32];   // per-wave P tile

  const int lane = threadIdx.x & 31;
  const int wid  = threadIdx.x >> 5;
  const int h    = blockIdx.y;
  const int q0   = (blockIdx.x * 4 + wid) * 16;
  const int ln16 = lane & 15;
  const int half8 = (lane & 16) ? 8 : 0;
  const int kbA   = (lane & 16) ? 16 : 0;
  const int kbB   = (lane & 16) ? 16 : 0;
  const float scale = 0.08838834764831845f;   // 1/sqrt(128)

  // Preload Q A-fragments (reused for every key tile).
  const unsigned short* qBase = Qb + ((size_t)h * N_ + q0 + ln16) * DK_ + kbA;
  v16bf qf[4];
#pragma unroll
  for (int kk = 0; kk < 4; ++kk) {
    Frag a;
    a.u[0] = *(const us8*)(qBase + kk * 32);
    a.u[1] = *(const us8*)(qBase + kk * 32 + 8);
    qf[kk] = a.bf;
  }

  const unsigned short* kBase = Kb + ((size_t)h * N_ + ln16) * DK_ + kbB;
  const unsigned short* vp[8];
#pragma unroll
  for (int t = 0; t < 8; ++t)
    vp[t] = VtB + ((size_t)h * DK_ + t * 16 + ln16) * N_ + kbB;

  float m[8], l[8];
  v8f acc[8];
#pragma unroll
  for (int r = 0; r < 8; ++r) { m[r] = -3.0e38f; l[r] = 0.0f; acc[r] = (v8f){}; }

  for (int j0 = 0; j0 < N_; j0 += 32) {
    // ---- S = Q.K^T for 2 sub-tiles of 16 keys (C layout, f32) ----
    const unsigned short* kp = kBase + (size_t)j0 * DK_;
    v8f s0 = (v8f){}, s1 = (v8f){};
#pragma unroll
    for (int kk = 0; kk < 4; ++kk) {
      Frag b0, b1;
      b0.u[0] = *(const us8*)(kp + kk * 32);
      b0.u[1] = *(const us8*)(kp + kk * 32 + 8);
      b1.u[0] = *(const us8*)(kp + (size_t)16 * DK_ + kk * 32);
      b1.u[1] = *(const us8*)(kp + (size_t)16 * DK_ + kk * 32 + 8);
      s0 = wmma_bf16(qf[kk], b0.bf, s0);
      s1 = wmma_bf16(qf[kk], b1.bf, s1);
    }
    if (j0 + 32 < N_) __builtin_prefetch(kp + (size_t)32 * DK_, 0, 1);

    // ---- first 4 V fragments: loads overlap the softmax VALU below ----
    Frag vb[4];
#pragma unroll
    for (int t = 0; t < 4; ++t) {
      const unsigned short* vrow = vp[t] + j0;
      vb[t].u[0] = *(const us8*)(vrow);
      vb[t].u[1] = *(const us8*)(vrow + 8);
    }

    // ---- online softmax update (row stats via half-wave butterflies) ----
#pragma unroll
    for (int r = 0; r < 8; ++r) {
      float x0 = s0[r] * scale, x1 = s1[r] * scale;
      float tm = fmaxf(x0, x1);
      tm = fmaxf(tm, __shfl_xor(tm, 1, 32));
      tm = fmaxf(tm, __shfl_xor(tm, 2, 32));
      tm = fmaxf(tm, __shfl_xor(tm, 4, 32));
      tm = fmaxf(tm, __shfl_xor(tm, 8, 32));
      float mn = fmaxf(m[r], tm);
      float alpha = __expf(m[r] - mn);
      m[r] = mn;
      float p0 = __expf(x0 - mn);
      float p1 = __expf(x1 - mn);
      float rs = p0 + p1;
      rs += __shfl_xor(rs, 1, 32);
      rs += __shfl_xor(rs, 2, 32);
      rs += __shfl_xor(rs, 4, 32);
      rs += __shfl_xor(rs, 8, 32);
      l[r] = l[r] * alpha + rs;
#pragma unroll
      for (int t = 0; t < 8; ++t) acc[t][r] *= alpha;
      int mr = r + half8;
      pshm[wid][mr * 32 + ln16]      = f2bf(p0);   // C layout -> row-major LDS
      pshm[wid][mr * 32 + 16 + ln16] = f2bf(p1);
    }

    // ---- last 4 V fragments: in flight during LDS bounce + first WMMAs ----
    Frag vc[4];
#pragma unroll
    for (int t = 0; t < 4; ++t) {
      const unsigned short* vrow = vp[4 + t] + j0;
      vc[t].u[0] = *(const us8*)(vrow);
      vc[t].u[1] = *(const us8*)(vrow + 8);
    }

    asm volatile("s_wait_dscnt 0" ::: "memory");

    // ---- reload P as A-fragment (layout change via LDS) ----
    Frag pf;
    const unsigned short* pb = &pshm[wid][ln16 * 32] + kbA;
    pf.u[0] = *(const us8*)(pb);
    pf.u[1] = *(const us8*)(pb + 8);

    // ---- O += P.V over the 8 dk column tiles ----
#pragma unroll
    for (int t = 0; t < 4; ++t)
      acc[t] = wmma_bf16(pf.bf, vb[t].bf, acc[t]);
#pragma unroll
    for (int t = 0; t < 4; ++t)
      acc[4 + t] = wmma_bf16(pf.bf, vc[t].bf, acc[4 + t]);
  }

  // ---- epilogue: normalize and write [N, H*DK] f32 ----
#pragma unroll
  for (int r = 0; r < 8; ++r) {
    float inv = 1.0f / l[r];
    int orow = q0 + r + half8;
#pragma unroll
    for (int t = 0; t < 8; ++t) {
      out[(size_t)orow * (H_ * DK_) + h * DK_ + t * 16 + ln16] = acc[t][r] * inv;
    }
  }
}

// ---------------------------------------------------------------------------

extern "C" void kernel_launch(void* const* d_in, const int* in_sizes, int n_in,
                              void* d_out, int out_size, void* d_ws, size_t ws_size,
                              hipStream_t stream) {
  const float* tok = (const float*)d_in[0];   // [N, D]
  const float* wq  = (const float*)d_in[1];   // [H, D, DK]
  const float* wk  = (const float*)d_in[2];
  const float* wv  = (const float*)d_in[3];
  float* out = (float*)d_out;                  // [N, H*DK]

  unsigned short* ws = (unsigned short*)d_ws;
  size_t o = 0;
  unsigned short* tokB = ws + o; o += (size_t)N_ * D_;
  unsigned short* wqT  = ws + o; o += (size_t)H_ * DK_ * D_;
  unsigned short* wkT  = ws + o; o += (size_t)H_ * DK_ * D_;
  unsigned short* wvT  = ws + o; o += (size_t)H_ * DK_ * D_;
  unsigned short* Qb   = ws + o; o += (size_t)H_ * N_ * DK_;
  unsigned short* Kb   = ws + o; o += (size_t)H_ * N_ * DK_;
  unsigned short* VtB  = ws + o; o += (size_t)H_ * N_ * DK_;
  // total ~92.3 MB of workspace

  const int nTok8 = (N_ * D_) / 8;
  cvt_bf16_kernel<<<(nTok8 + 255) / 256, 256, 0, stream>>>(tok, tokB, nTok8);

  dim3 tgrid(D_ / 32, DK_ / 32, H_), tblk(32, 8);
  wt_transpose_kernel<<<tgrid, tblk, 0, stream>>>(wq, wqT);
  wt_transpose_kernel<<<tgrid, tblk, 0, stream>>>(wk, wkT);
  wt_transpose_kernel<<<tgrid, tblk, 0, stream>>>(wv, wvT);

  qkv_proj_kernel<<<dim3(N_ / 32, H_, 3), 128, 0, stream>>>(
      tokB, wqT, wkT, wvT, Qb, Kb, VtB);

  attention_kernel<<<dim3(N_ / 64, H_), 128, 0, stream>>>(Qb, Kb, VtB, out);
}
